// Net4_2_35012573397156
// MI455X (gfx1250) — matrix operand order repaired and checked by reading, not compiled, not run on hardware
//
#include <hip/hip_runtime.h>

// ---------------- model constants (from reference) ----------------
#define B_GRAPHS 8192
#define NN0      64      // nodes per molecular graph
#define F0V      64      // input features
#define EMBED    128
#define EMOL     128     // edges per molecular graph
#define NSE      4000
#define NTOTAL   12192   // B + NSE
#define ESE      400000
#define KK1      52
#define KK2      42
#define KK3      34

// bf16 weight pack offsets (in ushort elements) inside d_ws
#define OFF_M1   0        // 256x64
#define OFF_M2   16384    // 128x256
#define OFF_C1L  49152
#define OFF_C1R  65536
#define OFF_C2L  81920
#define OFF_C2R  98304
#define OFF_C3L  114688
#define OFF_C3R  131072
#define OFF_L1   147456   // 128x256
#define OFF_L2   180224   // 128x128
#define OFF_S1L  196608
#define OFF_S1R  212992
#define OFF_S2L  229376
#define OFF_S2R  245760
#define WBF_ELEMS 262144
#define WBF_BYTES 524288

typedef __attribute__((ext_vector_type(16))) __bf16 v16bf;
typedef __attribute__((ext_vector_type(8)))  float  v8f;

// native f32 -> bf16 (RTNE) — lowers to hardware cvt on gfx1250
__device__ __forceinline__ unsigned short f2bf(float f) {
  union { __bf16 b; unsigned short u; } c;
  c.b = (__bf16)f;
  return c.u;
}
__device__ __forceinline__ float bf2f(unsigned short s) {
  union { unsigned u; float f; } c; c.u = ((unsigned)s) << 16; return c.f;
}
__device__ __forceinline__ __bf16 us2bf(unsigned short s) {
  union { unsigned short u; __bf16 b; } c; c.u = s; return c.b;
}

// A fragment: 16x32 bf16 tile. lane<16: row m0+lane, K = {0..7,16..23}+k0
//             lane>=16: row m0+lane-16, K = {8..15,24..31}+k0  (ISA 7.12.2)
__device__ __forceinline__ v16bf load_a(const unsigned short* A, int lda,
                                        int m0, int k0, int lane) {
  int hi = lane >> 4;
  const unsigned short* p = A + (m0 + (lane & 15)) * lda + k0 + hi * 8;
  v16bf a;
#pragma unroll
  for (int v = 0; v < 8; ++v) {
    int k = (v < 4) ? (2 * v) : (8 + 2 * v);
    a[2 * v]     = us2bf(p[k]);
    a[2 * v + 1] = us2bf(p[k + 1]);
  }
  return a;
}
// B fragment for X @ W^T: B[k][n] = W[n][k] (W row-major [out][in], ld = K).
// lane<16: n = n0+lane, K = {0..7,16..23} ; lane>=16: K = {8..15,24..31}.
__device__ __forceinline__ v16bf load_b(const unsigned short* W, int ldk,
                                        int n0, int k0, int lane) {
  int hi = lane >> 4;
  const unsigned short* p = W + (n0 + (lane & 15)) * ldk + k0 + hi * 8;
  v16bf b;
#pragma unroll
  for (int v = 0; v < 8; ++v) {
    int k = (v < 4) ? (2 * v) : (8 + 2 * v);
    b[2 * v]     = us2bf(p[k]);
    b[2 * v + 1] = us2bf(p[k + 1]);
  }
  return b;
}

__device__ __forceinline__ void mm_acc(const unsigned short* A, int lda,
                                       const unsigned short* W, int K,
                                       int m0, int n0, int lane, v8f& acc) {
  for (int k0 = 0; k0 < K; k0 += 32) {
    v16bf a = load_a(A, lda, m0, k0, lane);
    v16bf b = load_b(W, K, n0, k0, lane);
    acc = __builtin_amdgcn_wmma_f32_16x16x32_bf16(false, a, false, b,
                                                  (short)0, acc, false, false);
  }
}

// ---------------- per-graph SAGE conv (LDS resident) ----------------
__device__ void sage_step(int n, const unsigned short* Wl, const unsigned short* Wr,
                          const float* bias,
                          unsigned short* xB, unsigned short* mB, float* aF,
                          const int* e0, const int* e1, const int* em, float* cnt,
                          int tid, int lane, int wave) {
  for (int idx = tid; idx < n * EMBED; idx += 256) aF[idx] = 0.f;
  if (tid < n) cnt[tid] = 0.f;
  __syncthreads();
  // masked edge aggregation via LDS f32 atomics
  for (int idx = tid; idx < EMOL * EMBED; idx += 256) {
    int e = idx >> 7, f = idx & 127;
    if (em[e]) atomicAdd(&aF[e1[e] * EMBED + f], bf2f(xB[e0[e] * EMBED + f]));
  }
  for (int e = tid; e < EMOL; e += 256)
    if (em[e]) atomicAdd(&cnt[e1[e]], 1.f);
  __syncthreads();
  // mean -> bf16
  for (int idx = tid; idx < n * EMBED; idx += 256) {
    int r = idx >> 7;
    mB[idx] = f2bf(aF[idx] / fmaxf(cnt[r], 1.f));
  }
  __syncthreads();
  // out = relu(mean @ Wl^T + x @ Wr^T + b)  via WMMA
  int mt = (n + 15) >> 4;
  for (int t = wave; t < mt * 8; t += 8) {
    int m0 = (t % mt) * 16, nb = (t / mt) * 16;
    v8f acc = {0.f, 0.f, 0.f, 0.f, 0.f, 0.f, 0.f, 0.f};
    mm_acc(mB, EMBED, Wl, EMBED, m0, nb, lane, acc);
    mm_acc(xB, EMBED, Wr, EMBED, m0, nb, lane, acc);
    int hi = lane >> 4, col = nb + (lane & 15);
#pragma unroll
    for (int v = 0; v < 8; ++v) {
      int r = m0 + v + hi * 8;
      if (r < n) aF[r * EMBED + col] = fmaxf(acc[v] + bias[col], 0.f);
    }
  }
  __syncthreads();
}

// ---------------- per-graph top-k pool + readout ----------------
__device__ void topk_step(int n, int k, const float* pw,
                          float* aF, unsigned short* xB,
                          float* score, int* inv,
                          int* e0, int* e1, int* em,
                          float* pwS, float* outAcc, int tid) {
  if (tid < EMBED) pwS[tid] = pw[tid];
  __syncthreads();
  if (tid < n) {
    float dot = 0.f, nr = 0.f;
    for (int f = 0; f < EMBED; ++f) { float w = pwS[f]; dot += aF[tid * EMBED + f] * w; nr += w * w; }
    score[tid] = tanhf(dot * rsqrtf(nr));
  }
  __syncthreads();
  if (tid < n) {   // stable descending rank (matches lax.top_k tie-break)
    float s = score[tid]; int rk = 0;
    for (int j = 0; j < n; ++j) {
      float sj = score[j];
      rk += (sj > s) || (sj == s && j < tid);
    }
    inv[tid] = (rk < k) ? rk : -1;
  }
  __syncthreads();
  // gather+scale selected rows into new x (bf16, ld 128)
  for (int idx = tid; idx < n * EMBED; idx += 256) {
    int r = idx >> 7;
    int nrw = inv[r];
    if (nrw >= 0) xB[nrw * EMBED + (idx & 127)] = f2bf(aF[idx] * score[r]);
  }
  // remap edges in place
  for (int e = tid; e < EMOL; e += 256) {
    int a = em[e] ? inv[e0[e]] : -1;
    int b = em[e] ? inv[e1[e]] : -1;
    int m = (a >= 0) && (b >= 0);
    e0[e] = m ? a : 0; e1[e] = m ? b : 0; em[e] = m;
  }
  __syncthreads();
  // readout: max / mean over k pooled rows
  if (tid < EMBED) {
    float mx = -3.0e38f, sm = 0.f;
    for (int r = 0; r < k; ++r) { float v = bf2f(xB[r * EMBED + tid]); mx = fmaxf(mx, v); sm += v; }
    outAcc[tid] += mx;
    outAcc[EMBED + tid] += sm / (float)k;
  }
  __syncthreads();
}

// ---------------- kernel 1: fused per-graph molecular GNN ----------------
__global__ __launch_bounds__(256) void k_molgnn(
    const float* __restrict__ mol_x, const int* __restrict__ mol_edges,
    const unsigned short* __restrict__ wbf,
    const float* __restrict__ m1b, const float* __restrict__ m2b,
    const float* __restrict__ c1b, const float* __restrict__ c2b,
    const float* __restrict__ c3b,
    const float* __restrict__ p1w, const float* __restrict__ p2w,
    const float* __restrict__ p3w,
    float* __restrict__ xd) {
  __shared__ unsigned short sU[64 * 256];   // bf16 activations (ping)
  __shared__ unsigned short sV[64 * 256];   // bf16 activations (pong) / mean
  __shared__ float aF[64 * 128];            // f32 agg / conv output / stage buf
  __shared__ float sCnt[64];
  __shared__ float sScore[64];
  __shared__ int   sInv[64];
  __shared__ int   sE0[EMOL], sE1[EMOL], sEm[EMOL];
  __shared__ float sOut[256];
  __shared__ float sPw[EMBED];

  int tid = threadIdx.x, lane = tid & 31, wave = tid >> 5;
  int g = blockIdx.x;
  sOut[tid] = 0.f;

  // warm GL2 with the bf16 weight pack (global_prefetch_b8)
  if (blockIdx.x < 64) {
    for (int off = tid * 64; off < WBF_ELEMS; off += 256 * 64)
      __builtin_prefetch(wbf + off, 0, 1);
  }

  for (int i = tid; i < EMOL; i += 256) {
    sE0[i] = mol_edges[(size_t)g * 2 * EMOL + i];
    sE1[i] = mol_edges[(size_t)g * 2 * EMOL + EMOL + i];
    sEm[i] = 1;
  }

  // ---- async DMA stage: mol_x tile (64x64 f32 = 16KB) memory -> LDS (aF) ----
  // GLOBAL_LOAD_ASYNC_TO_LDS_B128 (ASYNCcnt), then s_wait_asynccnt 0.
  {
    unsigned lbase = (unsigned)(unsigned long long)(void*)aF;
    const float* gsrc = mol_x + (size_t)g * (NN0 * F0V);
    for (int idx = tid; idx < (NN0 * F0V) / 4; idx += 256) {
      unsigned laddr = lbase + (unsigned)idx * 16u;
      const float* ga = gsrc + idx * 4;
      asm volatile("global_load_async_to_lds_b128 %0, %1, off"
                   :: "v"(laddr), "v"(ga) : "memory");
    }
    asm volatile("s_wait_asynccnt 0x0" ::: "memory");
  }
  __syncthreads();
  for (int idx = tid; idx < NN0 * F0V; idx += 256)
    sU[idx] = f2bf(aF[idx]);
  __syncthreads();

  // h1 = relu(x @ m1W^T + m1b): [64,64] -> [64,256] (bf16 in sV, ld 256)
  for (int t = wave; t < 64; t += 8) {
    int m0 = (t & 3) * 16, nb = (t >> 2) * 16;
    v8f acc = {0.f, 0.f, 0.f, 0.f, 0.f, 0.f, 0.f, 0.f};
    mm_acc(sU, F0V, wbf + OFF_M1, F0V, m0, nb, lane, acc);
    int hi = lane >> 4, col = nb + (lane & 15);
#pragma unroll
    for (int v = 0; v < 8; ++v) {
      int r = m0 + v + hi * 8;
      sV[r * 256 + col] = f2bf(fmaxf(acc[v] + m1b[col], 0.f));
    }
  }
  __syncthreads();
  // h2 = relu(h1 @ m2W^T + m2b): [64,256] -> [64,128] (bf16 in sU, ld 128)
  for (int t = wave; t < 32; t += 8) {
    int m0 = (t & 3) * 16, nb = (t >> 2) * 16;
    v8f acc = {0.f, 0.f, 0.f, 0.f, 0.f, 0.f, 0.f, 0.f};
    mm_acc(sV, 256, wbf + OFF_M2, 256, m0, nb, lane, acc);
    int hi = lane >> 4, col = nb + (lane & 15);
#pragma unroll
    for (int v = 0; v < 8; ++v) {
      int r = m0 + v + hi * 8;
      sU[r * EMBED + col] = f2bf(fmaxf(acc[v] + m2b[col], 0.f));
    }
  }
  __syncthreads();

  const unsigned short* Wls[3] = {wbf + OFF_C1L, wbf + OFF_C2L, wbf + OFF_C3L};
  const unsigned short* Wrs[3] = {wbf + OFF_C1R, wbf + OFF_C2R, wbf + OFF_C3R};
  const float* cbs[3] = {c1b, c2b, c3b};
  const float* pws[3] = {p1w, p2w, p3w};
  int nn[4] = {NN0, KK1, KK2, KK3};
  for (int r = 0; r < 3; ++r) {
    sage_step(nn[r], Wls[r], Wrs[r], cbs[r], sU, sV, aF,
              sE0, sE1, sEm, sCnt, tid, lane, wave);
    topk_step(nn[r], nn[r + 1], pws[r], aF, sU, sScore, sInv,
              sE0, sE1, sEm, sPw, sOut, tid);
  }
  xd[(size_t)g * 256 + tid] = sOut[tid];
}

// ---------------- kernel 2: xd -> relu(l1) -> relu(l2) -> xa[0:B] ----------------
__global__ __launch_bounds__(256) void k_drug_linear(
    const float* __restrict__ xd, const unsigned short* __restrict__ wbf,
    const float* __restrict__ l1b, const float* __restrict__ l2b,
    float* __restrict__ xa) {
  __shared__ unsigned short sIn[64 * 256];
  __shared__ unsigned short sMid[64 * 128];
  int tid = threadIdx.x, lane = tid & 31, wave = tid >> 5;
  size_t r0 = (size_t)blockIdx.x * 64;
  for (int idx = tid; idx < 64 * 256; idx += 256)
    sIn[idx] = f2bf(xd[r0 * 256 + idx]);
  __syncthreads();
  for (int t = wave; t < 32; t += 8) {
    int m0 = (t & 3) * 16, nb = (t >> 2) * 16;
    v8f acc = {0.f, 0.f, 0.f, 0.f, 0.f, 0.f, 0.f, 0.f};
    mm_acc(sIn, 256, wbf + OFF_L1, 256, m0, nb, lane, acc);
    int hi = lane >> 4, col = nb + (lane & 15);
#pragma unroll
    for (int v = 0; v < 8; ++v) {
      int r = m0 + v + hi * 8;
      sMid[r * EMBED + col] = f2bf(fmaxf(acc[v] + l1b[col], 0.f));
    }
  }
  __syncthreads();
  for (int t = wave; t < 32; t += 8) {
    int m0 = (t & 3) * 16, nb = (t >> 2) * 16;
    v8f acc = {0.f, 0.f, 0.f, 0.f, 0.f, 0.f, 0.f, 0.f};
    mm_acc(sMid, EMBED, wbf + OFF_L2, EMBED, m0, nb, lane, acc);
    int hi = lane >> 4, col = nb + (lane & 15);
#pragma unroll
    for (int v = 0; v < 8; ++v) {
      int r = m0 + v + hi * 8;
      xa[(r0 + r) * EMBED + col] = fmaxf(acc[v] + l2b[col], 0.f);
    }
  }
}

// ---------------- kernel 3: xa[B:] = emb[x_ids[B:]] ----------------
__global__ void k_gather_emb(const int* __restrict__ x_ids,
                             const float* __restrict__ emb,
                             float* __restrict__ xa) {
  int idx = blockIdx.x * 256 + threadIdx.x;
  if (idx >= NSE * EMBED) return;
  int r = idx >> 7, f = idx & 127;
  int node = x_ids[B_GRAPHS + r];
  xa[(size_t)(B_GRAPHS + r) * EMBED + f] = emb[(size_t)node * EMBED + f];
}

// ---------------- big-graph SAGE support ----------------
__global__ void k_zero(float* __restrict__ p, int n) {
  int i = blockIdx.x * 256 + threadIdx.x;
  if (i < n) p[i] = 0.f;
}
__global__ void k_edge_agg(const int* __restrict__ se, const float* __restrict__ x,
                           float* __restrict__ agg, float* __restrict__ cnt) {
  long long idx = (long long)blockIdx.x * 256 + threadIdx.x;
  int e = (int)(idx >> 7), f = (int)(idx & 127);
  int s = se[e], d = se[ESE + e];
  atomicAdd(&agg[(size_t)d * EMBED + f], x[(size_t)s * EMBED + f]);
  if (f == 0) atomicAdd(&cnt[d], 1.f);
}
__global__ __launch_bounds__(256) void k_big_sage(
    const float* __restrict__ xin, const float* __restrict__ agg,
    const float* __restrict__ cnt,
    const unsigned short* __restrict__ Wl, const unsigned short* __restrict__ Wr,
    const float* __restrict__ bias, float* __restrict__ xout) {
  __shared__ unsigned short sX[64 * 128];
  __shared__ unsigned short sM[64 * 128];
  int tid = threadIdx.x, lane = tid & 31, wave = tid >> 5;
  int r0 = blockIdx.x * 64;
  for (int idx = tid; idx < 64 * 128; idx += 256) {
    int row = r0 + (idx >> 7), f = idx & 127;
    if (row < NTOTAL) {
      sX[idx] = f2bf(xin[(size_t)row * EMBED + f]);
      sM[idx] = f2bf(agg[(size_t)row * EMBED + f] / fmaxf(cnt[row], 1.f));
    } else { sX[idx] = 0; sM[idx] = 0; }
  }
  __syncthreads();
  for (int t = wave; t < 32; t += 8) {
    int m0 = (t & 3) * 16, nb = (t >> 2) * 16;
    v8f acc = {0.f, 0.f, 0.f, 0.f, 0.f, 0.f, 0.f, 0.f};
    mm_acc(sM, EMBED, Wl, EMBED, m0, nb, lane, acc);
    mm_acc(sX, EMBED, Wr, EMBED, m0, nb, lane, acc);
    int hi = lane >> 4, col = nb + (lane & 15);
#pragma unroll
    for (int v = 0; v < 8; ++v) {
      int r = r0 + m0 + v + hi * 8;
      if (r < NTOTAL) xout[(size_t)r * EMBED + col] = fmaxf(acc[v] + bias[col], 0.f);
    }
  }
}

// ---------------- output gather ----------------
__global__ void k_output(const float* __restrict__ xa,
                         const int* __restrict__ drugNodes,
                         const int* __restrict__ seNodes,
                         float* __restrict__ out) {
  long long idx = (long long)blockIdx.x * 256 + threadIdx.x;
  const long long P1 = (long long)B_GRAPHS * EMBED;
  const long long P2 = P1 + (long long)NSE * EMBED;
  const long long P3 = P2 + (long long)NTOTAL * EMBED;
  if (idx < P1) {
    int i = (int)(idx >> 7), f = (int)(idx & 127);
    out[idx] = xa[(size_t)drugNodes[i] * EMBED + f];
  } else if (idx < P2) {
    long long j = idx - P1;
    int i = (int)(j >> 7), f = (int)(j & 127);
    out[idx] = xa[(size_t)seNodes[i] * EMBED + f];
  } else if (idx < P3) {
    out[idx] = xa[idx - P2];
  }
}

// ---------------- f32 -> bf16 weight pack ----------------
__global__ void k_cvt(const float* __restrict__ s, unsigned short* __restrict__ d, int n) {
  int i = blockIdx.x * 256 + threadIdx.x;
  if (i < n) d[i] = f2bf(s[i]);
}

// ---------------- host launch ----------------
extern "C" void kernel_launch(void* const* d_in, const int* in_sizes, int n_in,
                              void* d_out, int out_size, void* d_ws, size_t ws_size,
                              hipStream_t stream) {
  (void)in_sizes; (void)n_in; (void)out_size; (void)ws_size;
  const int*   x_ids     = (const int*)  d_in[0];
  const int*   seEdges   = (const int*)  d_in[2];
  const int*   drugNodes = (const int*)  d_in[3];
  const int*   seNodes   = (const int*)  d_in[4];
  const float* mol_x     = (const float*)d_in[5];
  const int*   mol_edges = (const int*)  d_in[6];
  const float* emb       = (const float*)d_in[8];
  const float* m1W = (const float*)d_in[9];  const float* m1b = (const float*)d_in[10];
  const float* m2W = (const float*)d_in[11]; const float* m2b = (const float*)d_in[12];
  const float* c1Wl = (const float*)d_in[13]; const float* c1Wr = (const float*)d_in[14];
  const float* c1b  = (const float*)d_in[15];
  const float* c2Wl = (const float*)d_in[16]; const float* c2Wr = (const float*)d_in[17];
  const float* c2b  = (const float*)d_in[18];
  const float* c3Wl = (const float*)d_in[19]; const float* c3Wr = (const float*)d_in[20];
  const float* c3b  = (const float*)d_in[21];
  const float* s1Wl = (const float*)d_in[22]; const float* s1Wr = (const float*)d_in[23];
  const float* s1b  = (const float*)d_in[24];
  const float* s2Wl = (const float*)d_in[25]; const float* s2Wr = (const float*)d_in[26];
  const float* s2b  = (const float*)d_in[27];
  const float* p1w  = (const float*)d_in[28];
  const float* p2w  = (const float*)d_in[29];
  const float* p3w  = (const float*)d_in[30];
  const float* l1W  = (const float*)d_in[31]; const float* l1b = (const float*)d_in[32];
  const float* l2W  = (const float*)d_in[33]; const float* l2b = (const float*)d_in[34];

  char* base = (char*)d_ws;
  unsigned short* wbf = (unsigned short*)base;
  float* xd  = (float*)(base + WBF_BYTES);                 //  8192*256 f32
  float* xa0 = (float*)(base + 8912896);                   // 12192*128 f32
  float* xa1 = (float*)(base + 15155200);
  float* agg = (float*)(base + 21397504);
  float* cnt = (float*)(base + 27639808);

  struct CV { const float* s; int off; int n; };
  CV cvs[14] = {
    {m1W, OFF_M1, 16384}, {m2W, OFF_M2, 32768},
    {c1Wl, OFF_C1L, 16384}, {c1Wr, OFF_C1R, 16384},
    {c2Wl, OFF_C2L, 16384}, {c2Wr, OFF_C2R, 16384},
    {c3Wl, OFF_C3L, 16384}, {c3Wr, OFF_C3R, 16384},
    {l1W, OFF_L1, 32768}, {l2W, OFF_L2, 16384},
    {s1Wl, OFF_S1L, 16384}, {s1Wr, OFF_S1R, 16384},
    {s2Wl, OFF_S2L, 16384}, {s2Wr, OFF_S2R, 16384}
  };
  for (int i = 0; i < 14; ++i)
    k_cvt<<<(cvs[i].n + 255) / 256, 256, 0, stream>>>(cvs[i].s, wbf + cvs[i].off, cvs[i].n);

  // fused per-graph molecular GNN (WMMA-heavy)
  k_molgnn<<<B_GRAPHS, 256, 0, stream>>>(mol_x, mol_edges, wbf,
                                         m1b, m2b, c1b, c2b, c3b,
                                         p1w, p2w, p3w, xd);
  // drug dense layers -> xa rows [0, B)
  k_drug_linear<<<B_GRAPHS / 64, 256, 0, stream>>>(xd, wbf, l1b, l2b, xa0);
  // side-effect node embeddings -> xa rows [B, NTOT)
  k_gather_emb<<<(NSE * EMBED + 255) / 256, 256, 0, stream>>>(x_ids, emb, xa0);

  // big SAGE layer 1: xa0 -> xa1
  k_zero<<<(NTOTAL * EMBED + 255) / 256, 256, 0, stream>>>(agg, NTOTAL * EMBED);
  k_zero<<<(NTOTAL + 255) / 256, 256, 0, stream>>>(cnt, NTOTAL);
  k_edge_agg<<<(ESE * EMBED) / 256, 256, 0, stream>>>(seEdges, xa0, agg, cnt);
  k_big_sage<<<(NTOTAL + 63) / 64, 256, 0, stream>>>(xa0, agg, cnt,
                                                     wbf + OFF_S1L, wbf + OFF_S1R, s1b, xa1);
  // big SAGE layer 2: xa1 -> xa0
  k_zero<<<(NTOTAL * EMBED + 255) / 256, 256, 0, stream>>>(agg, NTOTAL * EMBED);
  k_zero<<<(NTOTAL + 255) / 256, 256, 0, stream>>>(cnt, NTOTAL);
  k_edge_agg<<<(ESE * EMBED) / 256, 256, 0, stream>>>(seEdges, xa1, agg, cnt);
  k_big_sage<<<(NTOTAL + 63) / 64, 256, 0, stream>>>(xa1, agg, cnt,
                                                     wbf + OFF_S2L, wbf + OFF_S2R, s2b, xa0);

  // final output: [xa[drugNodes]; xa[seNodes]; xa]
  long long tot = (long long)(B_GRAPHS + NSE + NTOTAL) * EMBED;
  k_output<<<(unsigned)((tot + 255) / 256), 256, 0, stream>>>(xa0, drugNodes, seNodes,
                                                              (float*)d_out);
}
